// MambaRegressorINT16_30983894073634
// MI455X (gfx1250) — matrix operand order repaired and checked by reading, not compiled, not run on hardware
//
#include <hip/hip_runtime.h>
#include <hip/hip_bf16.h>

// ---- dims (match reference) ----
// B=32, L=1024, DIN=8, D=128, KK=16, S=8, K=127, NB=4, F=24, Q=256
#define BB   32
#define LL   1024
#define DIN  8
#define DD   128
#define KP   127      // patch tokens
#define NBLK 4
#define FF   24
#define NOUT (DD * FF)        // 3072
#define KFLAT (DD * KP)       // 16256
#define MROWS (BB * KP)       // 4064

typedef __attribute__((ext_vector_type(16))) _Float16 v16h;
typedef __attribute__((ext_vector_type(8)))  float    v8f;

#define DEV __device__ __forceinline__

// ---- Q8.8 fixed-point helpers (round-half-even like jnp.round) ----
DEV float q88f(float x) {
  float q = rintf(x * 256.0f);
  q = fminf(32767.0f, fmaxf(-32768.0f, q));
  return q * 0.00390625f;   // /256
}
DEV float sigf(float x) { return 1.0f / (1.0f + __expf(-x)); }
DEV float silu_lut(float x) { float xq = q88f(x); return q88f(xq * sigf(xq)); }
DEV float sig_lut(float x)  { return q88f(sigf(q88f(x))); }

// =====================================================================
// 1) pointwise projection: xp[b,d,l] = q88( sum_c q88(x[b,l,c])*q88(wp[d,c]) + q88(bp[d]) )
// =====================================================================
__global__ void proj_kernel(const float* __restrict__ x,
                            const float* __restrict__ wp,
                            const float* __restrict__ bp,
                            float* __restrict__ xp) {
  int tid = blockIdx.x * blockDim.x + threadIdx.x;
  if (tid >= BB * DD * LL) return;
  int l = tid & (LL - 1);
  int d = (tid >> 10) & (DD - 1);
  int b = tid >> 17;
  float acc = q88f(bp[d]);
#pragma unroll
  for (int c = 0; c < DIN; ++c)
    acc += q88f(x[((size_t)(b * LL + l)) * DIN + c]) * q88f(wp[d * DIN + c]);
  xp[tid] = q88f(acc);
}

// =====================================================================
// 2) WMMA GEMM, fully specialized via templates:  C[m,n] = sum_q A[m,q]*B[q,n]
//    AMODE: 0 row-major A[m*lda+q]
//           1 patch-conv im2col gather from xp[B][D][L] (m=(b,k), q=(c,t));
//             both 8-element segments stay inside one c => two float4 loads
//    WT:    0 -> B[q,n]=W[q*ldw+n] (scalar loads)   1 -> B[q,n]=W[n*ldw+q]
//    QA/QW: apply q88 to A / W
//    POST:  0 raw (split-K partials), 1 q88(acc+q88(bias)),
//           2 q88(acc+q88(bias)) + pe_scale*pe[(m%127)*ldc+n],
//           3 silu_lut(acc), 4 C += acc
//    One wave computes one 16x16 tile; v_wmma_f32_16x16x32_f16, f32 accum.
// =====================================================================
template <int AMODE, int QA, int WT, int QW, int POST, int SPLITK>
__global__ __launch_bounds__(128) void gemm_wmma(
    const float* __restrict__ A, int lda,
    const float* __restrict__ W, int ldw,
    const float* __restrict__ bias,
    const float* __restrict__ pe,
    const float* __restrict__ pe_scale,
    float* __restrict__ C, int ldc,
    int M, int N, int Kq) {
  const int lane = threadIdx.x & 31;
  const int wave = threadIdx.x >> 5;
  const int tilesN = N >> 4;
  const int tilesM = M >> 4;
  const int tile = blockIdx.x * (blockDim.x >> 5) + wave;
  if (tile >= tilesM * tilesN) return;
  const int tm = tile / tilesN;
  const int tn = tile % tilesN;

  const int kchunk = Kq / SPLITK;
  const int kbeg = (SPLITK > 1) ? (int)blockIdx.y * kchunk : 0;
  const int kend = kbeg + kchunk;

  const int mrow = tm * 16 + (lane & 15);       // A row this lane contributes
  const int hk   = (lane < 16) ? 0 : 8;         // K sub-offset per 16x32 f16 A layout

  // loop-invariant pieces of the AMODE=1 im2col address
  const int pb = (AMODE == 1) ? (mrow / KP) : 0;
  const int pk = (AMODE == 1) ? (mrow % KP) : 0;
  const float* pbase = (AMODE == 1) ? (A + (size_t)pb * DD * LL + pk * 8) : nullptr;

  v8f acc = {};
  for (int q0 = kbeg; q0 < kend; q0 += 32) {
    const int qb = q0 + lane;                   // weight prefetch row offset
    if (WT == 1)                                // stream next weight block
      __builtin_prefetch(&W[(size_t)(tn * 16) * ldw + qb + 32], 0, 1);

    // ---- A fragment: 16 f16 per lane, two 8-float contiguous segments ----
    const float4 *plo, *phi;
    if (AMODE == 0) {
      plo = (const float4*)(A + (size_t)mrow * lda + q0 + hk);
      phi = (const float4*)(A + (size_t)mrow * lda + q0 + 16 + hk);
    } else {
      int ql = q0 + hk, qh = q0 + 16 + hk;      // both ql%16, qh%16 in {0,8}
      plo = (const float4*)(pbase + (size_t)(ql >> 4) * LL + (ql & 15));
      phi = (const float4*)(pbase + (size_t)(qh >> 4) * LL + (qh & 15));
    }
    float4 l0 = plo[0], l1 = plo[1], h0 = phi[0], h1 = phi[1];
    float av[16] = {l0.x, l0.y, l0.z, l0.w, l1.x, l1.y, l1.z, l1.w,
                    h0.x, h0.y, h0.z, h0.w, h1.x, h1.y, h1.z, h1.w};
    v16h a;
#pragma unroll
    for (int j = 0; j < 16; ++j)
      a[j] = (_Float16)(QA ? q88f(av[j]) : av[j]);

    // ---- B fragment: column N = lane%16, 16 K values in the same order as A ----
    v16h bf;
    if (WT == 0) {
#pragma unroll
      for (int j = 0; j < 16; ++j) {
        float w = W[(size_t)(q0 + hk + j + (j & 8)) * ldw + tn * 16 + (lane & 15)];
        bf[j] = (_Float16)(QW ? q88f(w) : w);
      }
    } else {
#pragma unroll
      for (int j = 0; j < 16; ++j) {
        float w = W[(size_t)(tn * 16 + (lane & 15)) * ldw + q0 + hk + j + (j & 8)];
        if (QW) w = q88f(w);
        bf[j] = (_Float16)w;
      }
    }

    acc = __builtin_amdgcn_wmma_f32_16x16x32_f16(false, a, false, bf,
                                                 (short)0, acc, false, false);
  }

  // ---- store C (VGPR r: M = base+r, N = lane%16) ----
  const int n = tn * 16 + (lane & 15);
  const int mbase = tm * 16 + ((lane < 16) ? 0 : 8);
  float* Cs = C + ((SPLITK > 1) ? (size_t)blockIdx.y * (size_t)M * ldc : 0);
  // one division for the whole epilogue; rows are consecutive so wrap by compare
  int kpe = (POST == 2) ? (mbase % KP) : 0;
  float psc = (POST == 2) ? pe_scale[0] : 0.0f;
#pragma unroll
  for (int r = 0; r < 8; ++r) {
    int m = mbase + r;
    float v = acc[r];
    size_t o = (size_t)m * ldc + n;
    if (SPLITK > 1 || POST == 0) {
      Cs[o] = v;
    } else if (POST == 1) {
      Cs[o] = q88f(v + q88f(bias[n]));
    } else if (POST == 2) {
      int kk = kpe + r;
      kk = (kk >= KP) ? kk - KP : kk;
      Cs[o] = q88f(v + q88f(bias[n])) + psc * pe[(size_t)kk * ldc + n];
    } else if (POST == 3) {
      Cs[o] = silu_lut(v);
    } else {
      Cs[o] += v;
    }
  }
}

// =====================================================================
// 3) LayerNorm over last dim (128), one block per row, LDS tree reduce
// =====================================================================
__global__ void ln_kernel(const float* __restrict__ X,
                          const float* __restrict__ g,
                          const float* __restrict__ bta,
                          float* __restrict__ Y) {
  __shared__ float red[DD];
  int row = blockIdx.x, t = threadIdx.x;
  float v = X[(size_t)row * DD + t];
  red[t] = v; __syncthreads();
  for (int s = DD / 2; s > 0; s >>= 1) { if (t < s) red[t] += red[t + s]; __syncthreads(); }
  float mean = red[0] * (1.0f / DD);
  __syncthreads();
  float dv = v - mean;
  red[t] = dv * dv; __syncthreads();
  for (int s = DD / 2; s > 0; s >>= 1) { if (t < s) red[t] += red[t + s]; __syncthreads(); }
  float var = red[0] * (1.0f / DD);
  Y[(size_t)row * DD + t] = dv * rsqrtf(var + 1e-5f) * g[t] + bta[t];
}

// =====================================================================
// 4) repack + quantize for the flat layer, div-free grid:
//    grid=(KP, BB), threads=DD:  aflat[b, d*127+k] = q88(hn[(b*127+k)*128+d])
// =====================================================================
__global__ void repack_kernel(const float* __restrict__ hn,
                              float* __restrict__ aflat) {
  int k = blockIdx.x;       // 0..126
  int b = blockIdx.y;       // 0..31
  int d = threadIdx.x;      // 0..127
  aflat[(size_t)b * KFLAT + (size_t)d * KP + k] =
      q88f(hn[((size_t)(b * KP + k)) * DD + d]);
}

// =====================================================================
// 5) dt projection: lam[b,t,d] = sig_lut( sum_k u[b,k,d]*w_dt[t,k] + b_dt[t] )
//    div-free grid: grid=(KP, BB), threads=DD
// =====================================================================
__global__ void dt_kernel(const float* __restrict__ u,
                          const float* __restrict__ wdt,
                          const float* __restrict__ bdt,
                          float* __restrict__ lam) {
  int t = blockIdx.x;       // 0..126
  int b = blockIdx.y;       // 0..31
  int d = threadIdx.x;      // 0..127
  float acc = bdt[t];
  for (int k = 0; k < KP; ++k)
    acc += u[((size_t)(b * KP + k)) * DD + d] * wdt[t * KP + k];
  lam[((size_t)(b * KP + t)) * DD + d] = sig_lut(acc);
}

// =====================================================================
// 6) SSM scan + gate: one thread per (b,d) chain; yg = scan(lam,u) * g
// =====================================================================
__global__ void scan_kernel(const float* __restrict__ lam,
                            const float* __restrict__ u,
                            const float* __restrict__ g,
                            float* __restrict__ yg) {
  int tid = blockIdx.x * blockDim.x + threadIdx.x;
  if (tid >= BB * DD) return;
  int b = tid >> 7, d = tid & (DD - 1);
  float s = 0.0f;
  for (int t = 0; t < KP; ++t) {
    size_t idx = ((size_t)(b * KP + t)) * DD + d;
    float l = lam[idx];
    s = l * s + (1.0f - l) * u[idx];
    yg[idx] = s * g[idx];
  }
}

// =====================================================================
// 7) split-K reduce + bias + q88 for the final flat layer
//    grid=(BB*NOUT/256); n = tid % NOUT is a power-of-2-free modulus only
//    if NOUT weren't 3072; use div-free 2D grid instead.
// =====================================================================
__global__ void finalize_kernel(const float* __restrict__ part,
                                const float* __restrict__ bflat,
                                float* __restrict__ yf) {
  int n = blockIdx.x * blockDim.x + threadIdx.x;   // 0..NOUT-1 (grid.x = NOUT/256)
  int b = blockIdx.y;                              // 0..BB-1
  size_t o = (size_t)b * NOUT + n;
  float s = part[o] + part[o + (size_t)BB * NOUT] + part[o + 2 * (size_t)BB * NOUT]
          + part[o + 3 * (size_t)BB * NOUT];
  yf[o] = q88f(s + q88f(bflat[n]));
}

// =====================================================================
// 8) head: out[b,j,f] = q88( sum_d q88(yf[b,d,f])*q88(wh[j,d]) + q88(bh[j]) )
// =====================================================================
__global__ void head_kernel(const float* __restrict__ yf,
                            const float* __restrict__ wh,
                            const float* __restrict__ bh,
                            float* __restrict__ out) {
  int tid = blockIdx.x * blockDim.x + threadIdx.x;
  if (tid >= BB * 2 * FF) return;
  int f = tid % FF;
  int j = (tid / FF) & 1;
  int b = tid / (2 * FF);
  float acc = q88f(bh[j]);
  for (int d = 0; d < DD; ++d)
    acc += q88f(yf[(size_t)b * NOUT + d * FF + f]) * q88f(wh[j * DD + d]);
  out[tid] = q88f(acc);
}

// =====================================================================
extern "C" void kernel_launch(void* const* d_in, const int* in_sizes, int n_in,
                              void* d_out, int out_size, void* d_ws, size_t ws_size,
                              hipStream_t stream) {
  (void)in_sizes; (void)n_in; (void)out_size; (void)ws_size;

  const float* x       = (const float*)d_in[0];
  const float* w_proj  = (const float*)d_in[1];
  const float* b_proj  = (const float*)d_in[2];
  const float* w_patch = (const float*)d_in[3];
  const float* b_patch = (const float*)d_in[4];
  const float* pe      = (const float*)d_in[5];
  const float* pe_sc   = (const float*)d_in[6];
  const float* ln_g    = (const float*)d_in[7];
  const float* ln_b    = (const float*)d_in[8];
  const float* w_in    = (const float*)d_in[9];
  const float* w_gate  = (const float*)d_in[10];
  const float* w_out   = (const float*)d_in[11];
  const float* w_dt    = (const float*)d_in[12];
  const float* b_dt    = (const float*)d_in[13];
  const float* lnf_g   = (const float*)d_in[14];
  const float* lnf_b   = (const float*)d_in[15];
  const float* w_flat  = (const float*)d_in[16];
  const float* b_flat  = (const float*)d_in[17];
  const float* w_head  = (const float*)d_in[18];
  const float* b_head  = (const float*)d_in[19];
  float* out = (float*)d_out;

  // workspace layout (floats); total ~7.81M floats (~31.2 MB), L2-resident
  float* ws   = (float*)d_ws;
  float* xp   = ws;                                   // 32*128*1024 (dead after patch conv)
  float* h    = xp  + (size_t)BB * DD * LL;           // 4064*128
  float* hn   = h   + (size_t)MROWS * DD;
  float* u    = hn  + (size_t)MROWS * DD;
  float* g    = u   + (size_t)MROWS * DD;
  float* lam  = g   + (size_t)MROWS * DD;
  float* yg   = lam + (size_t)MROWS * DD;
  float* part = yg  + (size_t)MROWS * DD;             // 4*32*3072
  float* yf   = part + (size_t)4 * BB * NOUT;         // 32*3072
  float* aflat = xp;                                  // reuse xp: 32*16256 floats

  // 1) pointwise projection (B,L,DIN) -> xp (B,D,L)
  proj_kernel<<<(BB * DD * LL) / 256, 256, 0, stream>>>(x, w_proj, b_proj, xp);

  // 2) patch conv as WMMA GEMM (im2col in A-loader) + bias + q88 + pos-emb
  //    M=4064, N=128, K=2048 -> 2032 tiles, 4 waves/block
  gemm_wmma<1, 0, 1, 1, 2, 1><<<dim3(508, 1), 128, 0, stream>>>(
      xp, 0, w_patch, DD * 16, b_patch, pe, pe_sc, h, DD, MROWS, DD, DD * 16);

  // 3) NB mamba-ish blocks
  for (int i = 0; i < NBLK; ++i) {
    ln_kernel<<<MROWS, DD, 0, stream>>>(h, ln_g + i * DD, ln_b + i * DD, hn);
    // u = silu_lut(hn @ w_in)
    gemm_wmma<0, 0, 0, 0, 3, 1><<<dim3(508, 1), 128, 0, stream>>>(
        hn, DD, w_in + (size_t)i * DD * DD, DD,
        nullptr, nullptr, nullptr, u, DD, MROWS, DD, DD);
    // g = silu_lut(hn @ w_gate)
    gemm_wmma<0, 0, 0, 0, 3, 1><<<dim3(508, 1), 128, 0, stream>>>(
        hn, DD, w_gate + (size_t)i * DD * DD, DD,
        nullptr, nullptr, nullptr, g, DD, MROWS, DD, DD);
    // lam = sig_lut(u . w_dt + b_dt)
    dt_kernel<<<dim3(KP, BB), DD, 0, stream>>>(
        u, w_dt + (size_t)i * KP * KP, b_dt + (size_t)i * KP, lam);
    scan_kernel<<<(BB * DD) / 256, 256, 0, stream>>>(lam, u, g, yg);
    // h += yg @ w_out
    gemm_wmma<0, 0, 0, 0, 4, 1><<<dim3(508, 1), 128, 0, stream>>>(
        yg, DD, w_out + (size_t)i * DD * DD, DD,
        nullptr, nullptr, nullptr, h, DD, MROWS, DD, DD);
  }

  // 4) final LN + repack/quantize A for the flat layer (div-free)
  ln_kernel<<<MROWS, DD, 0, stream>>>(h, lnf_g, lnf_b, hn);
  repack_kernel<<<dim3(KP, BB), DD, 0, stream>>>(hn, aflat);

  // 5) flat layer: bandwidth-bound 200MB stream of w_flat.
  //    M=32, N=3072, K=16256; split-K=4 (chunk 4064 = 127*32) -> 1536 waves
  gemm_wmma<0, 0, 1, 1, 0, 4><<<dim3(96, 4), 128, 0, stream>>>(
      aflat, KFLAT, w_flat, KFLAT, nullptr, nullptr, nullptr,
      part, NOUT, BB, NOUT, KFLAT);
  finalize_kernel<<<dim3(NOUT / 256, BB), 256, 0, stream>>>(part, b_flat, yf);

  // 6) head conv1x1 -> (32,2,24)
  head_kernel<<<(BB * 2 * FF + 255) / 256, 256, 0, stream>>>(yf, w_head, b_head, out);
}
